// MultiHeadSparseAttention_85100482003493
// MI455X (gfx1250) — compile-verified, matrix operand order checked by reading, not run on hardware
//
#include <hip/hip_runtime.h>
#include <hip/hip_bf16.h>

// Sizes from the reference
#define BB 4
#define SS 1024
#define DD 1024
#define HH 16
#define DP 64

typedef __attribute__((ext_vector_type(16))) _Float16 v16h;
typedef __attribute__((ext_vector_type(8)))  _Float16 v8h;
typedef __attribute__((ext_vector_type(8)))  float    v8f;
typedef __attribute__((ext_vector_type(4)))  float    v4f;

#define WMMA_F16(a, b, c) \
  __builtin_amdgcn_wmma_f32_16x16x32_f16(false, (a), false, (b), (short)0, (c), false, false)

// ---------------------------------------------------------------------------
// Fragment loaders, K-contiguous layouts only (ISA 7.12.2 layout):
// lane 0-15 -> row/col = lane, K = {0..7, 16..23}; lane 16-31 -> K = {8..15, 24..31}.
// f16 fragments are two 16-byte loads per lane, f32 fragments four.
// ---------------------------------------------------------------------------
__device__ inline v16h ldfragK(const _Float16* __restrict__ base, long ld, int lane) {
  const int row = lane & 15;
  const int kh8 = (lane >> 4) << 3;
  const _Float16* p = base + (long)row * ld + kh8;
  const v8h lo = *(const v8h*)p;
  const v8h hi = *(const v8h*)(p + 16);
  v16h f;
#pragma unroll
  for (int i = 0; i < 8; ++i) { f[i] = lo[i]; f[8 + i] = hi[i]; }
  return f;
}

__device__ inline v16h ldfragK_f32(const float* __restrict__ base, long ld, int lane) {
  const int row = lane & 15;
  const int kh8 = (lane >> 4) << 3;
  const float* p = base + (long)row * ld + kh8;
  const v4f a0 = *(const v4f*)p;
  const v4f a1 = *(const v4f*)(p + 4);
  const v4f b0 = *(const v4f*)(p + 16);
  const v4f b1 = *(const v4f*)(p + 20);
  v16h f;
#pragma unroll
  for (int i = 0; i < 4; ++i) {
    f[i]      = (_Float16)a0[i];
    f[4 + i]  = (_Float16)a1[i];
    f[8 + i]  = (_Float16)b0[i];
    f[12 + i] = (_Float16)b1[i];
  }
  return f;
}

// ---------------------------------------------------------------------------
// Kernel 0: transpose+convert 1024x1024 f32 -> f16 [N][K].
// z selects {wq, wk, wv, wo, G}. 32x32 tiles via LDS, block (32,8).
// ---------------------------------------------------------------------------
__global__ void tconv_kernel(const float* __restrict__ wq, const float* __restrict__ wk,
                             const float* __restrict__ wv, const float* __restrict__ wo,
                             const float* __restrict__ G,
                             _Float16* __restrict__ wqt, _Float16* __restrict__ wkt,
                             _Float16* __restrict__ wvt, _Float16* __restrict__ wot,
                             _Float16* __restrict__ Gt) {
  __shared__ float tile[32][33];
  const float* in; _Float16* out;
  switch (blockIdx.z) {
    case 0:  in = wq; out = wqt; break;
    case 1:  in = wk; out = wkt; break;
    case 2:  in = wv; out = wvt; break;
    case 3:  in = wo; out = wot; break;
    default: in = G;  out = Gt;  break;
  }
  const int tx = threadIdx.x, ty = threadIdx.y;
  const int r0 = blockIdx.y * 32, c0 = blockIdx.x * 32;
#pragma unroll
  for (int i = 0; i < 4; ++i)
    tile[ty * 4 + i][tx] = in[(long)(r0 + ty * 4 + i) * DD + c0 + tx];
  __syncthreads();
#pragma unroll
  for (int i = 0; i < 4; ++i)
    out[(long)(c0 + ty * 4 + i) * DD + r0 + tx] = (_Float16)tile[tx][ty * 4 + i];
}

// ---------------------------------------------------------------------------
// Kernel 1: QKV projection. grid = (N/64, M/128, 3), block = 256 (8 waves).
// Each wave: one 16-row M tile x 4 N tiles, A fragment reused 4x.
// q -> qh [B,H,S,DP] (K=dp contiguous for the scores GEMM);
// k,v -> kh_t/vh_t [B,H,DP,S] (K=s contiguous for kg / attn@V GEMMs),
// which makes their stores contiguous v8h b128 stores.
// ---------------------------------------------------------------------------
__global__ void proj_kernel(const float* __restrict__ q, const float* __restrict__ k,
                            const float* __restrict__ v,
                            const _Float16* __restrict__ wqt, const float* __restrict__ bq,
                            const _Float16* __restrict__ wkt, const float* __restrict__ bk,
                            const _Float16* __restrict__ wvt, const float* __restrict__ bv,
                            _Float16* __restrict__ qh, _Float16* __restrict__ kh_t,
                            _Float16* __restrict__ vh_t) {
  const int lane = threadIdx.x & 31;
  const int wave = threadIdx.x >> 5;
  const float* X; const _Float16* Wt; const float* bias; _Float16* Y;
  const int sel = blockIdx.z;
  if (sel == 0)      { X = q; Wt = wqt; bias = bq; Y = qh; }
  else if (sel == 1) { X = k; Wt = wkt; bias = bk; Y = kh_t; }
  else               { X = v; Wt = wvt; bias = bv; Y = vh_t; }

  const int m0 = blockIdx.y * 128 + wave * 16;   // 4096 rows / 128 per block
  const int n0 = blockIdx.x * 64;                // 1024 cols / 64 per block
  v8f acc[4] = {};
  for (int kb = 0; kb < DD; kb += 32) {
    const v16h a = ldfragK_f32(X + (long)m0 * DD + kb, DD, lane);
#pragma unroll
    for (int nt = 0; nt < 4; ++nt) {
      const v16h b = ldfragK(Wt + (long)(n0 + nt * 16) * DD + kb, DD, lane);
      acc[nt] = WMMA_F16(a, b, acc[nt]);
    }
  }
  const int mb = m0 + ((lane >> 4) << 3);        // 8 consecutive tokens
  const int bidx = mb >> 10, s0 = mb & 1023;
#pragma unroll
  for (int nt = 0; nt < 4; ++nt) {
    const int col = n0 + nt * 16 + (lane & 15);
    const int h = col >> 6, d = col & 63;
    const float bval = bias[col];
    const long bh = (long)bidx * HH + h;
    if (sel == 0) {
      // qh [bh][s][dp] : scattered 16-bit stores (stride DP)
#pragma unroll
      for (int r = 0; r < 8; ++r)
        Y[(bh * SS + (s0 + r)) * DP + d] = (_Float16)(acc[nt][r] + bval);
    } else {
      // kh_t / vh_t [bh][dp][s] : 8 contiguous f16 -> one b128 store
      v8h pk;
#pragma unroll
      for (int r = 0; r < 8; ++r) pk[r] = (_Float16)(acc[nt][r] + bval);
      *(v8h*)(Y + (bh * DP + d) * SS + s0) = pk;
    }
  }
}

// ---------------------------------------------------------------------------
// Kernel 2: kg_t[b,h][t][d] = (kh^T @ G)^T. grid = (S/128, B*H), block = 256.
// A = kh_t [d][s] (K contiguous), B = Gt [t][s] (K contiguous).
// Output stored [t][d] so the scores GEMM reads it K-contiguously; d is the
// fragment row dim -> contiguous v8h b128 store.
// ---------------------------------------------------------------------------
__global__ void kg_kernel(const _Float16* __restrict__ kh_t, const _Float16* __restrict__ Gt,
                          _Float16* __restrict__ kg_t) {
  const int lane = threadIdx.x & 31;
  const int wave = threadIdx.x >> 5;
  const int bh = blockIdx.y;
  const _Float16* A = kh_t + (long)bh * DP * SS;
  const int m0 = (wave & 3) * 16;                         // DP = 64 -> 4 M tiles
  const int nbase = blockIdx.x * 128 + (wave >> 2) * 64;  // 4 N tiles per wave
  v8f acc[4] = {};
  for (int kb = 0; kb < SS; kb += 32) {
    const v16h a = ldfragK(A + (long)m0 * SS + kb, SS, lane);
#pragma unroll
    for (int nt = 0; nt < 4; ++nt) {
      const v16h b = ldfragK(Gt + (long)(nbase + nt * 16) * SS + kb, SS, lane);
      acc[nt] = WMMA_F16(a, b, acc[nt]);
    }
  }
  const int mb = m0 + ((lane >> 4) << 3);                 // 8 consecutive d
#pragma unroll
  for (int nt = 0; nt < 4; ++nt) {
    const int t = nbase + nt * 16 + (lane & 15);
    v8h pk;
#pragma unroll
    for (int r = 0; r < 8; ++r) pk[r] = (_Float16)acc[nt][r];
    *(v8h*)(kg_t + ((long)bh * SS + t) * DP + mb) = pk;
  }
}

// ---------------------------------------------------------------------------
// Kernel 3: fused scores -> mask -> softmax -> attn@V.
// grid = (S/16, B*H), block = 256 (8 waves). 16x1024 score tile in LDS;
// attn_weights streamed out once with NT stores, mask streamed in with NT loads.
// ---------------------------------------------------------------------------
#define LDW 1032   // padded LDS row stride (floats); 4128 B, 16B aligned

__global__ void attn_kernel(const _Float16* __restrict__ qh, const _Float16* __restrict__ kg_t,
                            const _Float16* __restrict__ vh_t, const int* __restrict__ mask,
                            float* __restrict__ attn_w, _Float16* __restrict__ attnbuf) {
  extern __shared__ float smem[];
  float* srow = smem;                 // 16 * LDW floats
  float* vred = smem + 16 * LDW;      // 4 * 256 floats (split-K reduction)

  const int lane = threadIdx.x & 31;
  const int wave = threadIdx.x >> 5;
  const int bh = blockIdx.y;
  const int b = bh >> 4, h = bh & 15;
  const int row0 = blockIdx.x * 16;

  const _Float16* Aq = qh + ((long)bh * SS + row0) * DP;     // [s][dp]
  const _Float16* Bk = kg_t + (long)bh * SS * DP;            // [t][dp]

  const v16h a0 = ldfragK(Aq, DP, lane);        // K = 0..31 of depth
  const v16h a1 = ldfragK(Aq + 32, DP, lane);   // K = 32..63

  const int col = lane & 15;
  const int mb  = (lane >> 4) << 3;

  // ---- Phase 1: scores + mask into LDS ----
  for (int t = 0; t < 8; ++t) {
    const int n0 = wave * 128 + t * 16;
    const v16h b0 = ldfragK(Bk + (long)n0 * DP, DP, lane);
    const v16h b1 = ldfragK(Bk + (long)n0 * DP + 32, DP, lane);
    v8f acc = {};
    acc = WMMA_F16(a0, b0, acc);
    acc = WMMA_F16(a1, b1, acc);
    const int c = n0 + col;
#pragma unroll
    for (int r = 0; r < 8; ++r) {
      const int rg = row0 + mb + r;
      const int mk = __builtin_nontemporal_load(&mask[((long)b * SS + rg) * SS + c]);
      srow[(mb + r) * LDW + c] = acc[r] * 0.125f - 100000.0f * (float)mk;
    }
  }
  __syncthreads();

  // ---- Phase 2: row softmax (2 rows per wave), stream weights out once ----
  for (int rr = 0; rr < 2; ++rr) {
    const int r = wave * 2 + rr;
    float* row = srow + r * LDW;
    float mx = -3.4e38f;
    for (int c = lane; c < SS; c += 32) mx = fmaxf(mx, row[c]);
#pragma unroll
    for (int off = 16; off; off >>= 1) mx = fmaxf(mx, __shfl_xor(mx, off, 32));
    float sum = 0.0f;
    for (int c = lane; c < SS; c += 32) {
      const float e = __expf(row[c] - mx);
      row[c] = e; sum += e;
    }
#pragma unroll
    for (int off = 16; off; off >>= 1) sum += __shfl_xor(sum, off, 32);
    const float inv = 1.0f / sum;
    float* outw = attn_w + ((long)bh * SS + (row0 + r)) * SS;
    for (int c = lane; c < SS; c += 32) {
      const float w = row[c] * inv;
      row[c] = w;
      __builtin_nontemporal_store(w, &outw[c]);   // only HBM touch of weights
    }
  }
  __syncthreads();

  // ---- Phase 3: attn = weights @ V, split-K across wave pairs ----
  const _Float16* Bv = vh_t + (long)bh * DP * SS;   // [d][s], K contiguous
  const int tile  = wave & 3;                       // 4 col tiles -> DP = 64
  const int khalf = wave >> 2;                      // K split 0..511 / 512..1023
  const int n0 = tile * 16;
  v8f acc = {};
  for (int kk = 0; kk < 16; ++kk) {
    const int kb = khalf * 512 + kk * 32;
    const v16h a  = ldfragK_f32(srow + kb, LDW, lane);        // ds_load_b128 x4
    const v16h bf = ldfragK(Bv + (long)n0 * SS + kb, SS, lane);
    acc = WMMA_F16(a, bf, acc);
  }
  if (khalf == 1) {
    float* dst = vred + tile * 256 + lane * 8;
#pragma unroll
    for (int r = 0; r < 8; ++r) dst[r] = acc[r];
  }
  __syncthreads();
  if (khalf == 0) {
    const float* src = vred + tile * 256 + lane * 8;
#pragma unroll
    for (int r = 0; r < 8; ++r) {
      const float val = acc[r] + src[r];
      const int s = row0 + mb + r;
      attnbuf[((long)b * SS + s) * DD + h * DP + n0 + col] = (_Float16)val;
    }
  }
}

// ---------------------------------------------------------------------------
// Kernel 4: output projection. grid = (N/64, M/128), block = 256.
// A = attnbuf [token][D] (K contiguous), B = wot [n][k] (K contiguous).
// ---------------------------------------------------------------------------
__global__ void out_kernel(const _Float16* __restrict__ attnbuf, const _Float16* __restrict__ wot,
                           const float* __restrict__ bo, float* __restrict__ out) {
  const int lane = threadIdx.x & 31;
  const int wave = threadIdx.x >> 5;
  const int m0 = blockIdx.y * 128 + wave * 16;
  const int n0 = blockIdx.x * 64;
  v8f acc[4] = {};
  for (int kb = 0; kb < DD; kb += 32) {
    const v16h a = ldfragK(attnbuf + (long)m0 * DD + kb, DD, lane);
#pragma unroll
    for (int nt = 0; nt < 4; ++nt) {
      const v16h b = ldfragK(wot + (long)(n0 + nt * 16) * DD + kb, DD, lane);
      acc[nt] = WMMA_F16(a, b, acc[nt]);
    }
  }
  const int mb = m0 + ((lane >> 4) << 3);
#pragma unroll
  for (int nt = 0; nt < 4; ++nt) {
    const int c = n0 + nt * 16 + (lane & 15);
    const float bval = bo[c];
#pragma unroll
    for (int r = 0; r < 8; ++r)
      out[(long)(mb + r) * DD + c] = acc[nt][r] + bval;
  }
}

// ---------------------------------------------------------------------------
extern "C" void kernel_launch(void* const* d_in, const int* in_sizes, int n_in,
                              void* d_out, int out_size, void* d_ws, size_t ws_size,
                              hipStream_t stream) {
  const float* v  = (const float*)d_in[0];
  const float* q  = (const float*)d_in[1];
  const float* k  = (const float*)d_in[2];
  const int*   mask = (const int*)d_in[3];
  const float* G  = (const float*)d_in[4];
  const float* wq = (const float*)d_in[5];
  const float* bq = (const float*)d_in[6];
  const float* wk = (const float*)d_in[7];
  const float* bk = (const float*)d_in[8];
  const float* wv = (const float*)d_in[9];
  const float* bv = (const float*)d_in[10];
  const float* wo = (const float*)d_in[11];
  const float* bo = (const float*)d_in[12];

  float* out    = (float*)d_out;                   // [B,S,D]
  float* attn_w = out + (long)BB * SS * DD;        // [B,H,S,S]

  const long NTOK = (long)BB * SS * DD;            // 4M elements
  const long NMAT = (long)DD * DD;                 // 1M elements
  _Float16* qh      = (_Float16*)d_ws;
  _Float16* kh_t    = qh + NTOK;                   // [B,H,DP,S]
  _Float16* vh_t    = kh_t + NTOK;                 // [B,H,DP,S]
  _Float16* kg_t    = vh_t + NTOK;                 // [B*H, S, DP]
  _Float16* attnbuf = kg_t + NTOK;                 // [B, S, D]
  _Float16* wqt     = attnbuf + NTOK;              // [N][K] f16
  _Float16* wkt     = wqt + NMAT;
  _Float16* wvt     = wkt + NMAT;
  _Float16* wot     = wvt + NMAT;
  _Float16* Gt      = wot + NMAT;

  // 0) one-time transpose+convert of weights and G to f16 [N][K]
  tconv_kernel<<<dim3(32, 32, 5), dim3(32, 8), 0, stream>>>(
      wq, wk, wv, wo, G, wqt, wkt, wvt, wot, Gt);

  // 1) QKV projections
  proj_kernel<<<dim3(DD / 64, (BB * SS) / 128, 3), 256, 0, stream>>>(
      q, k, v, wqt, bq, wkt, bk, wvt, bv, qh, kh_t, vh_t);

  // 2) kg_t = (K^T @ G)^T per head
  kg_kernel<<<dim3(SS / 128, BB * HH), 256, 0, stream>>>(kh_t, Gt, kg_t);

  // 3) fused scores + mask + softmax + attn@V
  const size_t smem = (size_t)(16 * LDW + 4 * 256) * sizeof(float);  // ~70 KB
  attn_kernel<<<dim3(SS / 16, BB * HH), 256, smem, stream>>>(
      qh, kg_t, vh_t, mask, attn_w, attnbuf);

  // 4) output projection
  out_kernel<<<dim3(DD / 64, (BB * SS) / 128), 256, 0, stream>>>(attnbuf, wot, bo, out);
}
